// FalconAttention_71992241816205
// MI455X (gfx1250) — matrix operand order repaired and checked by reading, not compiled, not run on hardware
//
#include <hip/hip_runtime.h>
#include <hip/hip_bf16.h>
#include <math.h>

// ---------------------------------------------------------------------------
// Falcon attention block for MI455X (gfx1250), wave32 + WMMA f16->f32.
// Pipeline:
//   1) f32->f16 convert (weights pre-transposed to N-major for K-contiguous
//      fragment staging)
//   2) QKV GEMM  (WMMA, async global->LDS staging, BK=64)
//   3) RoPE + split (V stored transposed [d][s] for contiguous PV fragments)
//   4) flash attention (WMMA QK^T + PV, online softmax, wave32 shuffles)
//   5) dense projection GEMM (WMMA) -> f32 output
// ---------------------------------------------------------------------------

typedef _Float16 half_t;
typedef __attribute__((ext_vector_type(16))) _Float16 v16h;
typedef __attribute__((ext_vector_type(8)))  _Float16 v8h;
typedef __attribute__((ext_vector_type(8)))  float    v8f;

#define NH    32
#define NKV   8
#define NREP  4
#define HD    64
#define S_LEN 2048
#define BATCH 2
#define HID   2048
#define EQKV  3072             // 32*64 + 2*8*64
#define MROWS (S_LEN * BATCH)  // 4096

// ---------------------------------------------------------------------------
// CDNA5 async global->LDS copy (ASYNCcnt path), per CDNA5 ISA.
// GVS mode: mem_addr = SADDR(64) + VADDR(i32) ; LDS dest addr in VDST vgpr.
// ---------------------------------------------------------------------------
__device__ __forceinline__ void async_ld_b128(unsigned lds_byte_addr,
                                              int gbyte_off,
                                              const void* gbase_uniform) {
  asm volatile("global_load_async_to_lds_b128 %0, %1, %2"
               :
               : "v"(lds_byte_addr), "v"(gbyte_off), "s"(gbase_uniform)
               : "memory");
}
__device__ __forceinline__ void wait_asynccnt0() {
  asm volatile("s_wait_asynccnt 0x0" ::: "memory");
}
__device__ __forceinline__ unsigned lds_addr_of(const void* p) {
  // Generic pointer to LDS: low 32 bits carry the group-segment byte offset.
  return (unsigned)(unsigned long long)p;
}

// ---------------------------------------------------------------------------
// fp32 -> f16 conversion (plain)
// ---------------------------------------------------------------------------
__global__ void cvt_f32_f16(const float* __restrict__ in,
                            half_t* __restrict__ out, int n) {
  int i = blockIdx.x * blockDim.x + threadIdx.x;
  if (i < n) out[i] = (half_t)in[i];
}

// fp32 (K x N, row-major) -> f16 transposed (N x K, row-major)
__global__ void cvt_t_f32_f16(const float* __restrict__ in,
                              half_t* __restrict__ out, int K, int N) {
  long idx = (long)blockIdx.x * blockDim.x + threadIdx.x;
  if (idx >= (long)K * N) return;
  int k = (int)(idx / N);
  int n = (int)(idx % N);
  out[(size_t)n * K + k] = (half_t)in[idx];
}

// ---------------------------------------------------------------------------
// Tiled WMMA GEMM: C(MxN) = A(MxK) * B^T(NxK both row-major), f16 in, f32 acc.
// Block: 256 threads = 8 waves. Block tile 128x128, K-step 64.
// Wave tile 64x32 = 4x2 grid of 16x16 WMMA tiles, 16 WMMAs per K-step.
// A and B tiles staged via async global->LDS b128 transfers.
// ---------------------------------------------------------------------------
#define LDS_STRIDE 72   // 64 + 8 pad (f16 elems); row pitch 144 B (16B aligned)

template <bool OUT_HALF>
__global__ void __launch_bounds__(256)
gemm_wmma(const half_t* __restrict__ A, const half_t* __restrict__ Bt,
          void* __restrict__ Cout, int M, int N, int K) {
  __shared__ half_t As[128 * LDS_STRIDE];
  __shared__ half_t Bs[128 * LDS_STRIDE];

  const int t    = threadIdx.x;
  const int wave = t >> 5;
  const int lane = t & 31;
  const int lc   = lane & 15;          // col / row-in-tile selector
  const int lh   = lane >> 4;          // half-wave selector
  const int wm   = (wave >> 2) * 64;   // wave M offset in block tile
  const int wn   = (wave & 3) * 32;    // wave N offset in block tile
  const long blockM = (long)blockIdx.y * 128;
  const long blockN = (long)blockIdx.x * 128;

  v8f acc[4][2];
#pragma unroll
  for (int i = 0; i < 4; i++)
#pragma unroll
    for (int j = 0; j < 2; j++) acc[i][j] = {};

  // Staging map: each thread moves one 32-f16 (64 B) row segment of each tile.
  const int srow = t >> 1;             // 0..127
  const int sseg = (t & 1) * 32;       // 0 or 32 (f16 elems)

  const void* Ab = (const void*)(A + (size_t)blockM * K);    // uniform
  const void* Bb = (const void*)(Bt + (size_t)blockN * K);   // uniform

  const unsigned ldsA = lds_addr_of(&As[srow * LDS_STRIDE + sseg]);
  const unsigned ldsB = lds_addr_of(&Bs[srow * LDS_STRIDE + sseg]);

  for (int k0 = 0; k0 < K; k0 += 64) {
    // --- async stage A (M x 64) and B^T (N x 64) tiles into LDS ---
    const int goff = (srow * K + k0 + sseg) * 2;   // bytes
#pragma unroll
    for (int j = 0; j < 4; j++)
      async_ld_b128(ldsA + j * 16, goff + j * 16, Ab);
#pragma unroll
    for (int j = 0; j < 4; j++)
      async_ld_b128(ldsB + j * 16, goff + j * 16, Bb);
    wait_asynccnt0();
    __syncthreads();

    // --- 2 WMMA K-substeps over the 64-deep tile ---
#pragma unroll
    for (int ks = 0; ks < 2; ks++) {
      v16h af[4];
#pragma unroll
      for (int mt = 0; mt < 4; mt++) {
        const half_t* r = &As[(wm + mt * 16 + lc) * LDS_STRIDE + ks * 32];
        v8h lo = *(const v8h*)(r + lh * 8);        // K = lh*8 .. +7
        v8h hi = *(const v8h*)(r + 16 + lh * 8);   // K = 16+lh*8 .. +7
#pragma unroll
        for (int i = 0; i < 8; i++) { af[mt][i] = lo[i]; af[mt][8 + i] = hi[i]; }
      }
      v16h bf[2];
#pragma unroll
      for (int nt = 0; nt < 2; nt++) {
        const half_t* r = &Bs[(wn + nt * 16 + lc) * LDS_STRIDE + ks * 32];
        v8h lo = *(const v8h*)(r + lh * 8);
        v8h hi = *(const v8h*)(r + 16 + lh * 8);
#pragma unroll
        for (int i = 0; i < 8; i++) { bf[nt][i] = lo[i]; bf[nt][8 + i] = hi[i]; }
      }
#pragma unroll
      for (int mt = 0; mt < 4; mt++)
#pragma unroll
        for (int nt = 0; nt < 2; nt++)
          acc[mt][nt] = __builtin_amdgcn_wmma_f32_16x16x32_f16(
              false, af[mt], false, bf[nt], (short)0, acc[mt][nt], false, false);
    }
    __syncthreads();
  }

  // --- writeback (C layout: lane = col lc, rows lh*8 + r) ---
  const int rowoff = lh * 8;
#pragma unroll
  for (int mt = 0; mt < 4; mt++)
#pragma unroll
    for (int nt = 0; nt < 2; nt++)
#pragma unroll
      for (int r = 0; r < 8; r++) {
        size_t gr = (size_t)(blockM + wm + mt * 16 + rowoff + r);
        size_t gc = (size_t)(blockN + wn + nt * 16 + lc);
        float v = acc[mt][nt][r];
        if (OUT_HALF) ((half_t*)Cout)[gr * N + gc] = (half_t)v;
        else          ((float*)Cout)[gr * N + gc]  = v;
      }
}

// ---------------------------------------------------------------------------
// RoPE + split fused QKV (rows = s*B+b, cols = kv*(6*64)+slot*64+d) into
// Q[b][h][s][d], K[b][kv][s][d], and V TRANSPOSED: Vt[b][kv][d][s]   (f16)
// ---------------------------------------------------------------------------
__global__ void rope_split(const half_t* __restrict__ qkv,
                           half_t* __restrict__ Q, half_t* __restrict__ Ko,
                           half_t* __restrict__ Vt) {
  long idx = (long)blockIdx.x * blockDim.x + threadIdx.x;
  if (idx >= (long)MROWS * EQKV) return;
  int row = (int)(idx / EQKV);        // s*B + b
  int e   = (int)(idx % EQKV);
  int s = row / BATCH, b = row % BATCH;
  int kv   = e / (6 * HD);
  int rem  = e % (6 * HD);
  int slot = rem / HD;                // 0..3 Q heads, 4 K, 5 V
  int d    = rem % HD;

  float x = (float)qkv[idx];
  if (slot == 5) {
    Vt[(((size_t)b * NKV + kv) * HD + d) * S_LEN + s] = (half_t)x;
    return;
  }
  // RoPE (rotate-half form): theta_j = s * 10000^(-2j/64), j = d mod 32
  int j = d & 31;
  float inv = __powf(10000.f, -(float)(2 * j) * (1.0f / 64.0f));
  float ang = (float)s * inv;
  float c = __cosf(ang), sn = __sinf(ang);
  int dp = (d < 32) ? d + 32 : d - 32;
  float xp  = (float)qkv[(size_t)row * EQKV + (e - d) + dp];
  float rot = (d < 32) ? -xp : xp;
  float out = x * c + rot * sn;

  if (slot < 4) {
    int h = kv * NREP + slot;
    Q[(((size_t)b * NH + h) * S_LEN + s) * HD + d] = (half_t)out;
  } else {
    Ko[(((size_t)b * NKV + kv) * S_LEN + s) * HD + d] = (half_t)out;
  }
}

// ---------------------------------------------------------------------------
// Flash attention: one wave per (b, h, 16-row Q tile). 32 keys per iteration.
// Scores via 2x WMMA K=32; online softmax with width-16 shuffles; P routed
// through per-wave LDS tile (C-layout -> A-fragment layout); P*V via 4x WMMA
// with V read from the transposed layout (two contiguous 8-f16 runs per lane).
// ---------------------------------------------------------------------------
__global__ void __launch_bounds__(256)
flash_attn(const half_t* __restrict__ Q, const half_t* __restrict__ Km,
           const half_t* __restrict__ Vt, const unsigned char* __restrict__ smask,
           half_t* __restrict__ O) {
  __shared__ half_t Pl[8][16 * 32];   // per-wave P tile (f16)

  const int t    = threadIdx.x;
  const int wave = t >> 5;
  const int lane = t & 31;
  const int lc   = lane & 15;
  const int lh   = lane >> 4;
  const int rowoff = lh * 8;

  const int tile = blockIdx.x * 8 + wave;      // 0 .. 8191
  const int qt   = tile & 127;                 // q tile (16 rows each)
  const int h    = (tile >> 7) & 31;
  const int b    = tile >> 12;
  const int kvh  = h >> 2;                     // GQA group

  const half_t* Qb  = Q  + (((size_t)b * NH + h) * S_LEN + (size_t)qt * 16) * HD;
  const half_t* Kb  = Km + ((size_t)b * NKV + kvh) * S_LEN * HD;
  const half_t* Vtb = Vt + ((size_t)b * NKV + kvh) * (size_t)HD * S_LEN;
  const unsigned char* sm = smask + (size_t)b * S_LEN;

  // Q A-fragments (16x64 -> two K=32 fragments), live for the whole loop
  v16h qa[2];
#pragma unroll
  for (int f = 0; f < 2; f++) {
    const half_t* r = Qb + lc * HD + f * 32;
    v8h lo = *(const v8h*)(r + lh * 8);
    v8h hi = *(const v8h*)(r + 16 + lh * 8);
#pragma unroll
    for (int i = 0; i < 8; i++) { qa[f][i] = lo[i]; qa[f][8 + i] = hi[i]; }
  }

  v8f acc[4];
#pragma unroll
  for (int nf = 0; nf < 4; nf++) acc[nf] = {};
  float mr[8], ls[8];
#pragma unroll
  for (int r = 0; r < 8; r++) { mr[r] = -1e30f; ls[r] = 0.f; }

  half_t* P = Pl[wave];
  const float scale = 0.125f;      // 1/sqrt(64)
  const int nkb = (qt >> 1) + 1;   // 32-key blocks covering keys 0..qt*16+15

  for (int kb = 0; kb < nkb; kb++) {
    const int kbase = kb * 32;
    v8f sc[2];
#pragma unroll
    for (int hN = 0; hN < 2; hN++) {
      v8f s = {};
#pragma unroll
      for (int kf = 0; kf < 2; kf++) {
        // B fragment of K^T (64x16): lane holds key-column, 16 d-values
        const half_t* r =
            Kb + (size_t)(kbase + hN * 16 + lc) * HD + kf * 32;
        v8h lo = *(const v8h*)(r + lh * 8);
        v8h hi = *(const v8h*)(r + 16 + lh * 8);
        v16h bkf;
#pragma unroll
        for (int i = 0; i < 8; i++) { bkf[i] = lo[i]; bkf[8 + i] = hi[i]; }
        s = __builtin_amdgcn_wmma_f32_16x16x32_f16(
            false, qa[kf], false, bkf, (short)0, s, false, false);
      }
      // scale + causal + sequence mask
#pragma unroll
      for (int r = 0; r < 8; r++) {
        int rg = qt * 16 + rowoff + r;
        int cg = kbase + hN * 16 + lc;
        bool ok = (cg <= rg) && (sm[cg] != 0);
        s[r] = ok ? s[r] * scale : -1e30f;
      }
      sc[hN] = s;
    }

    // online softmax, per owned row (replicated across each 16-lane half)
#pragma unroll
    for (int r = 0; r < 8; r++) {
      float rm = fmaxf(sc[0][r], sc[1][r]);
#pragma unroll
      for (int off = 1; off < 16; off <<= 1)
        rm = fmaxf(rm, __shfl_xor(rm, off, 16));
      float mnew  = fmaxf(mr[r], rm);
      float alpha = __expf(mr[r] - mnew);
      mr[r] = mnew;
      float p0 = __expf(sc[0][r] - mnew);
      float p1 = __expf(sc[1][r] - mnew);
      float ps = p0 + p1;
#pragma unroll
      for (int off = 1; off < 16; off <<= 1) ps += __shfl_xor(ps, off, 16);
      ls[r] = ls[r] * alpha + ps;
#pragma unroll
      for (int nf = 0; nf < 4; nf++) acc[nf][r] *= alpha;
      // stash P (f16) in C-layout position; reread below in A-frag layout
      P[(rowoff + r) * 32 + lc]      = (half_t)p0;
      P[(rowoff + r) * 32 + 16 + lc] = (half_t)p1;
    }

    // P as A-fragment (16x32). Same-wave DS ordering guarantees visibility.
    v16h pf;
    {
      const half_t* r = P + lc * 32;
      v8h lo = *(const v8h*)(r + lh * 8);
      v8h hi = *(const v8h*)(r + 16 + lh * 8);
#pragma unroll
      for (int i = 0; i < 8; i++) { pf[i] = lo[i]; pf[8 + i] = hi[i]; }
    }

    // V B-fragments (32x64) from transposed V: two contiguous 8-f16 runs per lane
#pragma unroll
    for (int nf = 0; nf < 4; nf++) {
      const half_t* r = Vtb + (size_t)(nf * 16 + lc) * S_LEN + kbase;
      v8h lo = *(const v8h*)(r + lh * 8);
      v8h hi = *(const v8h*)(r + 16 + lh * 8);
      v16h vf;
#pragma unroll
      for (int i = 0; i < 8; i++) { vf[i] = lo[i]; vf[8 + i] = hi[i]; }
      acc[nf] = __builtin_amdgcn_wmma_f32_16x16x32_f16(
          false, pf, false, vf, (short)0, acc[nf], false, false);
    }
  }

  // normalize + write O in (S, B, NH*HD) layout (= GEMM2 A operand)
#pragma unroll
  for (int r = 0; r < 8; r++) {
    float inv = 1.0f / ls[r];
    int s = qt * 16 + rowoff + r;
    size_t base = ((size_t)s * BATCH + b) * HID + (size_t)h * HD;
#pragma unroll
    for (int nf = 0; nf < 4; nf++)
      O[base + nf * 16 + lc] = (half_t)(acc[nf][r] * inv);
  }
}

// ---------------------------------------------------------------------------
// Launch
// ---------------------------------------------------------------------------
extern "C" void kernel_launch(void* const* d_in, const int* in_sizes, int n_in,
                              void* d_out, int out_size, void* d_ws,
                              size_t ws_size, hipStream_t stream) {
  const float* hs            = (const float*)d_in[0];          // (S,B,H) f32
  const unsigned char* smask = (const unsigned char*)d_in[1];  // (B,S) bool
  const float* wqkv          = (const float*)d_in[2];          // (H,E) f32
  const float* wd            = (const float*)d_in[3];          // (H,H) f32
  float* out                 = (float*)d_out;                  // (S,B,H) f32

  half_t* ws  = (half_t*)d_ws;
  half_t* Xh  = ws;                                    //  4096*2048
  half_t* WqT = Xh  + (size_t)MROWS * HID;             //  3072*2048 (E x H)
  half_t* WdT = WqT + (size_t)HID * EQKV;              //  2048*2048 (O x H)
  half_t* QKV = WdT + (size_t)HID * HID;               //  4096*3072
  half_t* Qr  = QKV + (size_t)MROWS * EQKV;            //  B*NH*S*HD
  half_t* Kr  = Qr  + (size_t)BATCH * NH * S_LEN * HD; //  B*NKV*S*HD
  half_t* Vt  = Kr  + (size_t)BATCH * NKV * S_LEN * HD;//  B*NKV*HD*S
  half_t* O   = Vt  + (size_t)BATCH * NKV * S_LEN * HD;//  4096*2048

  // 1) convert inputs to f16; weights transposed to N-major
  {
    int n0 = MROWS * HID;
    cvt_f32_f16<<<(n0 + 255) / 256, 256, 0, stream>>>(hs, Xh, n0);
    int n1 = HID * EQKV;
    cvt_t_f32_f16<<<(n1 + 255) / 256, 256, 0, stream>>>(wqkv, WqT, HID, EQKV);
    int n2 = HID * HID;
    cvt_t_f32_f16<<<(n2 + 255) / 256, 256, 0, stream>>>(wd, WdT, HID, HID);
  }
  // 2) fused QKV GEMM (f16 out)
  gemm_wmma<true><<<dim3(EQKV / 128, MROWS / 128), 256, 0, stream>>>(
      Xh, WqT, QKV, MROWS, EQKV, HID);
  // 3) RoPE + split into head-major Q/K and transposed V
  {
    long n = (long)MROWS * EQKV;
    rope_split<<<(unsigned)((n + 255) / 256), 256, 0, stream>>>(QKV, Qr, Kr, Vt);
  }
  // 4) flash attention (8192 wave-tiles, 8 waves per block)
  flash_attn<<<(BATCH * NH * (S_LEN / 16)) / 8, 256, 0, stream>>>(
      Qr, Kr, Vt, smask, O);
  // 5) dense projection (f32 out -> d_out)
  gemm_wmma<false><<<dim3(HID / 128, MROWS / 128), 256, 0, stream>>>(
      O, WdT, out, MROWS, HID, HID);
}
